// PolicyStep_71347996721384
// MI455X (gfx1250) — matrix operand order repaired and checked by reading, not compiled, not run on hardware
//
#include <hip/hip_runtime.h>
#include <hip/hip_bf16.h>
#include <math.h>

// ---------------------------------------------------------------------------
// Problem constants
// ---------------------------------------------------------------------------
#define BB    4
#define TT    2048
#define VV    8192
#define DD    512
#define NOPS  8
#define HIDD  1024
#define POUT  (DD + NOPS + DD)      // 1032
#define MROWS (BB * TT)             // 8192

typedef __attribute__((ext_vector_type(16))) __bf16 v16bf;
typedef __attribute__((ext_vector_type(8)))  float  v8f;
typedef __attribute__((ext_vector_type(4)))  int    v4i_;

#define GLOBAL_AS __attribute__((address_space(1)))
#define LDS_AS    __attribute__((address_space(3)))

// ---------------------------------------------------------------------------
// CDNA5 async global->LDS staging (ASYNCcnt-tracked DMA), with safe fallback
// ---------------------------------------------------------------------------
#if defined(__has_builtin)
#  if __has_builtin(__builtin_amdgcn_global_load_async_to_lds_b128) && \
      __has_builtin(__builtin_amdgcn_s_wait_asynccnt)
#    define USE_ASYNC_LDS 1
#  endif
#endif
#ifndef USE_ASYNC_LDS
#  define USE_ASYNC_LDS 0
#endif

// stage one 16-byte chunk global -> LDS
__device__ __forceinline__ void stage16(const __bf16* g, __bf16* l) {
#if USE_ASYNC_LDS
    __builtin_amdgcn_global_load_async_to_lds_b128(
        (GLOBAL_AS v4i_*)g,
        (LDS_AS    v4i_*)l, 0, 0);
#else
    *(uint4*)l = *(const uint4*)g;
#endif
}

__device__ __forceinline__ void async_fence() {
#if USE_ASYNC_LDS
    __builtin_amdgcn_s_wait_asynccnt(0);
#endif
}

// round-to-nearest-even fp32 -> bf16
__device__ __forceinline__ __bf16 f2bf(float f) {
    union { float f; unsigned u; } v; v.f = f;
    unsigned r = v.u + 0x7FFFu + ((v.u >> 16) & 1u);
    unsigned short h = (unsigned short)(r >> 16);
    return __builtin_bit_cast(__bf16, h);
}

__device__ __forceinline__ v8f wmma_bf16(v16bf a, v16bf b, v8f c) {
    // D = A(16x32) * B(32x16) + C, fp32 accum
    return __builtin_amdgcn_wmma_f32_16x16x32_bf16(false, a, false, b,
                                                   (short)0, c, false, false);
}

__device__ __forceinline__ float gelu_exact(float x) {
    return 0.5f * x * (1.0f + erff(x * 0.70710678118654752f));
}

// ---------------------------------------------------------------------------
// RMSNorm over vocab dim, emit bf16
// ---------------------------------------------------------------------------
__global__ __launch_bounds__(256)
void rmsnorm_bf16_kernel(const float* __restrict__ x, __bf16* __restrict__ xn) {
    int m = blockIdx.x, t = threadIdx.x;
    const float* row = x + (size_t)m * VV;
    float ss = 0.0f;
    for (int i = t; i < VV; i += 256) { float v = row[i]; ss += v * v; }
    __shared__ float red[256];
    red[t] = ss; __syncthreads();
    for (int s = 128; s > 0; s >>= 1) {
        if (t < s) red[t] += red[t + s];
        __syncthreads();
    }
    float scale = rsqrtf(red[0] / (float)VV + 1.1920928955078125e-07f);
    __bf16* orow = xn + (size_t)m * VV;
    for (int i = t; i < VV; i += 256) orow[i] = f2bf(row[i] * scale);
}

// ---------------------------------------------------------------------------
// fp32 -> bf16 weight conversion (grid-stride)
// ---------------------------------------------------------------------------
__global__ __launch_bounds__(256)
void cvt_bf16_kernel(const float* __restrict__ in, __bf16* __restrict__ out, size_t n) {
    size_t i = (size_t)blockIdx.x * 256 + threadIdx.x;
    size_t stride = (size_t)gridDim.x * 256;
    for (; i < n; i += stride) out[i] = f2bf(in[i]);
}

// ---------------------------------------------------------------------------
// Generic bf16 WMMA GEMM:  C(M,N) = A(M,K) @ W(N,K)^T   [+bias][GELU][*scale]
// Block tile 128x64: 8 waves, each wave 16(M) x 64(N) = 4 accumulators.
// K-step 32; tiles staged via async global->LDS DMA.
// ---------------------------------------------------------------------------
__global__ __launch_bounds__(256)
void gemm_bf16_kernel(const __bf16* __restrict__ A, const __bf16* __restrict__ W,
                      const float* __restrict__ bias,
                      float* __restrict__ Cf, __bf16* __restrict__ Cbf,
                      const float* __restrict__ scale_ptr,
                      int M, int N, int K, int act) {
    __shared__ __bf16 As[128][40];
    __shared__ __bf16 Bs[64][40];
    int tid = threadIdx.x, lane = tid & 31, wid = tid >> 5;
    int bm = blockIdx.y * 128, bn = blockIdx.x * 64;
    int am = lane & 15, akg = lane >> 4;
    v8f acc[4] = {};

    for (int k0 = 0; k0 < K; k0 += 32) {
        { // stage A tile 128x32 (2 threads per row, 16 bf16 = 2 x b128 each)
            int row = tid >> 1, half = tid & 1;
            const __bf16* src = A + (size_t)(bm + row) * K + k0 + half * 16;
            stage16(src,     &As[row][half * 16]);
            stage16(src + 8, &As[row][half * 16 + 8]);
            if (k0 + 32 < K) __builtin_prefetch(src + 32, 0, 1); // global_prefetch_b8
        }
        { // stage B tile 64x32 (4 threads per row, 8 bf16 = 1 x b128 each), N-guarded
            int row = tid >> 2, seg = tid & 3;
            int gn = bn + row;
            if (gn < N)
                stage16(W + (size_t)gn * K + k0 + seg * 8, &Bs[row][seg * 8]);
        }
        async_fence();
        __syncthreads();
        v16bf af;
#pragma unroll
        for (int e = 0; e < 16; ++e)
            af[e] = As[wid * 16 + am][e + 16 * akg];
#pragma unroll
        for (int sub = 0; sub < 4; ++sub) {
            v16bf bfr;
#pragma unroll
            for (int e = 0; e < 16; ++e)
                bfr[e] = Bs[sub * 16 + am][e + 16 * akg];
            acc[sub] = wmma_bf16(af, bfr, acc[sub]);
        }
        __syncthreads();
    }

    float scl = scale_ptr ? scale_ptr[0] : 1.0f;
    int cm0 = bm + wid * 16 + akg * 8;
#pragma unroll
    for (int r = 0; r < 8; ++r) {
        int m = cm0 + r;
#pragma unroll
        for (int sub = 0; sub < 4; ++sub) {
            int n = bn + sub * 16 + am;
            if (n >= N) continue;
            float v = acc[sub][r];
            if (bias) v += bias[n];
            if (act == 1) v = gelu_exact(v);
            v *= scl;
            if (Cf)  Cf[(size_t)m * N + n] = v;
            if (Cbf) Cbf[(size_t)m * N + n] = f2bf(v);
        }
    }
}

// ---------------------------------------------------------------------------
// Anti-causal decay linear attention (flash-style, no TxT materialization)
//   mem[i,:] = sum_{j>i} (q_i . k_j) * decay^(j-i-1) * v[j,:]
//   h2 = h + mem_scale * mem      (fp32 + bf16 outputs)
// One block per (batch, 16-row i-tile). qkv packed per row: [q(512) k(512) v(512)]
// ---------------------------------------------------------------------------
__global__ __launch_bounds__(256)
void attn_decay_kernel(const __bf16* __restrict__ qkv, const float* __restrict__ hf,
                       float* __restrict__ h2f, __bf16* __restrict__ h2bf,
                       const float* __restrict__ decay_logit_p,
                       const float* __restrict__ mem_scale_p) {
    __shared__ __bf16 Vs[32][520];   // V tile: 32 j-rows x 512 d
    __shared__ __bf16 Pls[16][40];   // decayed score tile P: 16 i x 32 j
    int b  = blockIdx.x >> 7;        // T/16 = 128 i-tiles per batch
    int it = blockIdx.x & 127;
    int i0 = it * 16;
    size_t rowbase = (size_t)b * TT;
    int tid = threadIdx.x, lane = tid & 31, wid = tid >> 5;
    int am = lane & 15, akg = lane >> 4;

    float decay  = 1.0f / (1.0f + expf(-decay_logit_p[0]));
    float mscale = mem_scale_p[0];
    v8f macc[4] = {};

    for (int jt = i0 & ~31; jt < TT; jt += 32) {
        { // stage V tile (8 threads per row, 64 bf16 = 8 x b128 each)
            int row = tid >> 3, off = (tid & 7) * 64;
            const __bf16* src = qkv + (rowbase + jt + row) * 1536 + 1024 + off;
#pragma unroll
            for (int c = 0; c < 8; ++c) stage16(src + c * 8, &Vs[row][off + c * 8]);
        }
        async_fence();
        __syncthreads();

        if (wid < 2) { // waves 0,1 each build one 16x16 S subtile (K = D = 512)
            v8f sacc = {};
            for (int kk = 0; kk < DD; kk += 32) {
                v16bf qa, kb;
#pragma unroll
                for (int e = 0; e < 16; ++e)
                    qa[e] = qkv[(rowbase + i0 + am) * 1536 + kk + e + 16 * akg];
#pragma unroll
                for (int e = 0; e < 16; ++e)
                    kb[e] = qkv[(rowbase + jt + wid * 16 + am) * 1536 + 512 + kk + e + 16 * akg];
                sacc = wmma_bf16(qa, kb, sacc);
            }
#pragma unroll
            for (int r = 0; r < 8; ++r) { // decay mask, scatter to P (C-layout)
                int il = r + 8 * akg;
                int i  = i0 + il;
                int j  = jt + wid * 16 + am;
                int diff = j - i;
                float w = (diff > 0) ? powf(decay, (float)(diff - 1)) : 0.0f;
                Pls[il][wid * 16 + am] = f2bf(sacc[r] * w);
            }
        }
        __syncthreads();

        // all 8 waves: mem_tile += P(16x32) @ V(32x16) for their 4 d-subtiles
        v16bf pa, vb;
#pragma unroll
        for (int e = 0; e < 16; ++e)
            pa[e] = Pls[am][e + 16 * akg];
#pragma unroll
        for (int dtl = 0; dtl < 4; ++dtl) {
            int dbase = (wid * 4 + dtl) * 16;
#pragma unroll
            for (int e = 0; e < 16; ++e)
                vb[e] = Vs[e + 16 * akg][dbase + am];
            macc[dtl] = wmma_bf16(pa, vb, macc[dtl]);
        }
        __syncthreads();
    }

#pragma unroll
    for (int dtl = 0; dtl < 4; ++dtl) {
        int d = (wid * 4 + dtl) * 16 + am;
#pragma unroll
        for (int r = 0; r < 8; ++r) {
            size_t m = rowbase + i0 + r + 8 * akg;
            float v = hf[m * DD + d] + mscale * macc[dtl][r];
            h2f[m * DD + d]  = v;
            h2bf[m * DD + d] = f2bf(v);
        }
    }
}

// ---------------------------------------------------------------------------
// Policy head post-processing per row: softmax(512) fused into sel = h2*read_w,
// softmax(8) -> op_w, sigmoid(512) -> write_w
// ---------------------------------------------------------------------------
__global__ __launch_bounds__(256)
void policy_post_kernel(const float* __restrict__ pol, const float* __restrict__ h2,
                        __bf16* __restrict__ sel, float* __restrict__ opw,
                        float* __restrict__ writew) {
    int m = blockIdx.x, t = threadIdx.x;
    const float* p = pol + (size_t)m * POUT;
    __shared__ float red[256];

    float mx = -1e30f;
    for (int i = t; i < DD; i += 256) mx = fmaxf(mx, p[i]);
    red[t] = mx; __syncthreads();
    for (int s = 128; s > 0; s >>= 1) { if (t < s) red[t] = fmaxf(red[t], red[t + s]); __syncthreads(); }
    mx = red[0]; __syncthreads();

    float sum = 0.0f;
    for (int i = t; i < DD; i += 256) sum += expf(p[i] - mx);
    red[t] = sum; __syncthreads();
    for (int s = 128; s > 0; s >>= 1) { if (t < s) red[t] += red[t + s]; __syncthreads(); }
    float inv = 1.0f / red[0];

    for (int i = t; i < DD; i += 256) {
        float rw = expf(p[i] - mx) * inv;
        sel[(size_t)m * DD + i] = f2bf(h2[(size_t)m * DD + i] * rw);
    }
    if (t == 0) {
        float m8 = -1e30f;
        for (int n = 0; n < NOPS; ++n) m8 = fmaxf(m8, p[DD + n]);
        float e8[NOPS], s8 = 0.0f;
        for (int n = 0; n < NOPS; ++n) { e8[n] = expf(p[DD + n] - m8); s8 += e8[n]; }
        for (int n = 0; n < NOPS; ++n) opw[(size_t)m * NOPS + n] = e8[n] / s8;
    }
    for (int i = t; i < DD; i += 256)
        writew[(size_t)m * DD + i] = 1.0f / (1.0f + expf(-p[DD + NOPS + i]));
}

// ---------------------------------------------------------------------------
// OpBank: 8 simultaneous GEMM banks, activation mixture in registers.
// result = write_w * sum_n op_w[n] * act_n(sel @ Wops[n]^T)   -> bf16
// Block tile 128x16, each wave holds 8 accumulators (one per op bank).
// ---------------------------------------------------------------------------
__global__ __launch_bounds__(256)
void opbank_kernel(const __bf16* __restrict__ sel, const __bf16* __restrict__ wops,
                   const float* __restrict__ opw, const float* __restrict__ writew,
                   __bf16* __restrict__ result) {
    __shared__ __bf16 As[128][40];
    __shared__ __bf16 Bs[NOPS][16][40];
    int tid = threadIdx.x, lane = tid & 31, wid = tid >> 5;
    int bm = blockIdx.y * 128, bn = blockIdx.x * 16;
    int am = lane & 15, akg = lane >> 4;
    v8f acc[NOPS] = {};

    for (int k0 = 0; k0 < DD; k0 += 32) {
        {
            int row = tid >> 1, half = tid & 1;
            const __bf16* src = sel + (size_t)(bm + row) * DD + k0 + half * 16;
            stage16(src,     &As[row][half * 16]);
            stage16(src + 8, &As[row][half * 16 + 8]);
        }
        { // 8 banks x 16 rows x 32 k  (one thread per (op,row,half))
            int op = tid >> 5, row = (tid >> 1) & 15, half = tid & 1;
            const __bf16* src = wops + ((size_t)op * DD + bn + row) * DD + k0 + half * 16;
            stage16(src,     &Bs[op][row][half * 16]);
            stage16(src + 8, &Bs[op][row][half * 16 + 8]);
        }
        async_fence();
        __syncthreads();
        v16bf af;
#pragma unroll
        for (int e = 0; e < 16; ++e)
            af[e] = As[wid * 16 + am][e + 16 * akg];
#pragma unroll
        for (int n = 0; n < NOPS; ++n) {
            v16bf bfr;
#pragma unroll
            for (int e = 0; e < 16; ++e) bfr[e] = Bs[n][am][e + 16 * akg];
            acc[n] = wmma_bf16(af, bfr, acc[n]);
        }
        __syncthreads();
    }

    int cm0 = bm + wid * 16 + akg * 8;
#pragma unroll
    for (int r = 0; r < 8; ++r) {
        int m = cm0 + r;
        int col = bn + am;
        const float* ow = opw + (size_t)m * NOPS;
        float x, tacc = 0.0f;
        x = acc[0][r]; tacc += ow[0] * gelu_exact(x);
        x = acc[1][r]; tacc += ow[1] * fmaxf(x, 0.0f);
        x = acc[2][r]; { float rr = fmaxf(x, 0.0f); tacc += ow[2] * rr * rr; }
        x = acc[3][r]; tacc += ow[3] * x / (1.0f + expf(-x));
        x = acc[4][r]; tacc += ow[4] * tanhf(x);
        x = acc[5][r]; tacc += ow[5] / (1.0f + expf(-x));
        x = acc[6][r]; tacc += ow[6] * x;
        x = acc[7][r]; tacc -= ow[7] * x;
        result[(size_t)m * DD + col] = f2bf(tacc * writew[(size_t)m * DD + col]);
    }
}

// ---------------------------------------------------------------------------
// Host orchestration
// ---------------------------------------------------------------------------
extern "C" void kernel_launch(void* const* d_in, const int* in_sizes, int n_in,
                              void* d_out, int out_size, void* d_ws, size_t ws_size,
                              hipStream_t stream) {
    (void)in_sizes; (void)n_in; (void)out_size; (void)ws_size;
    const float* x      = (const float*)d_in[0];
    const float* Wc     = (const float*)d_in[1];
    const float* We     = (const float*)d_in[2];
    const float* Wq     = (const float*)d_in[3];
    const float* Wk     = (const float*)d_in[4];
    const float* Wv     = (const float*)d_in[5];
    const float* W1     = (const float*)d_in[6];
    const float* b1     = (const float*)d_in[7];
    const float* W2     = (const float*)d_in[8];
    const float* b2     = (const float*)d_in[9];
    const float* Wops   = (const float*)d_in[10];
    const float* decay  = (const float*)d_in[11];
    const float* mscale = (const float*)d_in[12];
    const float* ascale = (const float*)d_in[13];
    float* out = (float*)d_out;

    char* cur = (char*)d_ws;
    auto alloc = [&](size_t bytes) -> char* {
        char* p = cur; cur += (bytes + 255) & ~(size_t)255; return p;
    };
    __bf16* xn_bf   = (__bf16*)alloc((size_t)MROWS * VV * 2);
    __bf16* Wc_bf   = (__bf16*)alloc((size_t)DD * VV * 2);
    __bf16* We_bf   = (__bf16*)alloc((size_t)VV * DD * 2);
    __bf16* Wqkv_bf = (__bf16*)alloc((size_t)3 * DD * DD * 2);
    __bf16* W1_bf   = (__bf16*)alloc((size_t)HIDD * DD * 2);
    __bf16* W2_bf   = (__bf16*)alloc((size_t)POUT * HIDD * 2);
    __bf16* Wops_bf = (__bf16*)alloc((size_t)NOPS * DD * DD * 2);
    float*  h_f     = (float*) alloc((size_t)MROWS * DD * 4);
    __bf16* h_bf    = (__bf16*)alloc((size_t)MROWS * DD * 2);
    __bf16* qkv_bf  = (__bf16*)alloc((size_t)MROWS * 3 * DD * 2);
    float*  h2_f    = (float*) alloc((size_t)MROWS * DD * 4);
    __bf16* h2_bf   = (__bf16*)alloc((size_t)MROWS * DD * 2);
    __bf16* t1_bf   = (__bf16*)alloc((size_t)MROWS * HIDD * 2);
    float*  pol_f   = (float*) alloc((size_t)MROWS * POUT * 4);
    float*  opw_f   = (float*) alloc((size_t)MROWS * NOPS * 4);
    float*  ww_f    = (float*) alloc((size_t)MROWS * DD * 4);
    __bf16* sel_bf  = (__bf16*)alloc((size_t)MROWS * DD * 2);
    __bf16* res_bf  = (__bf16*)alloc((size_t)MROWS * DD * 2);

    // 1) RMSNorm -> bf16, weight conversions
    rmsnorm_bf16_kernel<<<MROWS, 256, 0, stream>>>(x, xn_bf);
    cvt_bf16_kernel<<<1024, 256, 0, stream>>>(Wc, Wc_bf, (size_t)DD * VV);
    cvt_bf16_kernel<<<1024, 256, 0, stream>>>(We, We_bf, (size_t)VV * DD);
    cvt_bf16_kernel<<<256, 256, 0, stream>>>(Wq, Wqkv_bf,                  (size_t)DD * DD);
    cvt_bf16_kernel<<<256, 256, 0, stream>>>(Wk, Wqkv_bf + (size_t)DD*DD,  (size_t)DD * DD);
    cvt_bf16_kernel<<<256, 256, 0, stream>>>(Wv, Wqkv_bf + (size_t)2*DD*DD,(size_t)DD * DD);
    cvt_bf16_kernel<<<256, 256, 0, stream>>>(W1, W1_bf,   (size_t)HIDD * DD);
    cvt_bf16_kernel<<<512, 256, 0, stream>>>(W2, W2_bf,   (size_t)POUT * HIDD);
    cvt_bf16_kernel<<<512, 256, 0, stream>>>(Wops, Wops_bf, (size_t)NOPS * DD * DD);

    // 2) compress: h = xn @ Wc^T   (fp32 + bf16)
    gemm_bf16_kernel<<<dim3(DD / 64, MROWS / 128), 256, 0, stream>>>(
        xn_bf, Wc_bf, nullptr, h_f, h_bf, nullptr, MROWS, DD, VV, 0);

    // 3) qkv = h @ [Wq;Wk;Wv]^T  (bf16, packed per row)
    gemm_bf16_kernel<<<dim3(3 * DD / 64, MROWS / 128), 256, 0, stream>>>(
        h_bf, Wqkv_bf, nullptr, nullptr, qkv_bf, nullptr, MROWS, 3 * DD, DD, 0);

    // 4) decay attention + residual: h2 = h + mem_scale * mem
    attn_decay_kernel<<<BB * (TT / 16), 256, 0, stream>>>(
        qkv_bf, h_f, h2_f, h2_bf, decay, mscale);

    // 5) MLP: t1 = gelu(h2 @ W1^T + b1);  pol = t1 @ W2^T + b2
    gemm_bf16_kernel<<<dim3(HIDD / 64, MROWS / 128), 256, 0, stream>>>(
        h2_bf, W1_bf, b1, nullptr, t1_bf, nullptr, MROWS, HIDD, DD, 1);
    gemm_bf16_kernel<<<dim3((POUT + 63) / 64, MROWS / 128), 256, 0, stream>>>(
        t1_bf, W2_bf, b2, pol_f, nullptr, nullptr, MROWS, POUT, HIDD, 0);

    // 6) policy heads: sel = h2*softmax, op_w, write_w
    policy_post_kernel<<<MROWS, 256, 0, stream>>>(pol_f, h2_f, sel_bf, opw_f, ww_f);

    // 7) OpBank mixture -> result (bf16)
    opbank_kernel<<<dim3(DD / 16, MROWS / 128), 256, 0, stream>>>(
        sel_bf, Wops_bf, opw_f, ww_f, res_bf);

    // 8) expand: out = (result @ We^T) * act_scale   (fp32 to d_out)
    gemm_bf16_kernel<<<dim3(VV / 64, MROWS / 128), 256, 0, stream>>>(
        res_bf, We_bf, nullptr, out, nullptr, ascale, MROWS, VV, DD, 0);
}
